// HANLayer_47321949667633
// MI455X (gfx1250) — compile-verified, matrix operand order checked by reading, not run on hardware
//
#include <hip/hip_runtime.h>
#include <hip/hip_bf16.h>
#include <math.h>

// ---------------- problem constants ----------------
static constexpr int PNUM = 4;       // meta-paths
static constexpr int RNUM = 3;       // relations
static constexpr int NN   = 60000;   // nodes per subgraph
static constexpr int NT   = 80000;   // global feature rows
static constexpr int EEN  = 960000;  // edges per graph
static constexpr int DIN  = 256;     // in features
static constexpr int OOUT = 256;     // out features
static constexpr int SN   = 4;       // slots: 3 rel + 1 root
static constexpr int CW   = SN * OOUT; // 1024 concatenated GEMM cols
static constexpr int HH   = 128;     // attention hidden
static constexpr int NREG = 50000;

typedef __bf16 bf16_t;
typedef __attribute__((ext_vector_type(16))) __bf16 v16bf;
typedef __attribute__((ext_vector_type(8)))  __bf16 v8bf;
typedef __attribute__((ext_vector_type(8)))  float  v8f;
typedef __attribute__((ext_vector_type(4)))  unsigned int v4u;
typedef __attribute__((ext_vector_type(8)))  int v8i;
typedef __attribute__((ext_vector_type(4)))  int v4i;

// LDS A-tile row pitch: 512B of data + 16B pad (TDM pad_interval=128dw, pad_amount=4dw)
// -> per-row bank shift of 4, conflict-light ds_load_b128 fragment reads.
static constexpr int APITCH = 528;

// Fragment slice for the 16x32 bf16 WMMA operand from a row-major [*,256] matrix.
// p already includes per-lane koff (0 or 8 elements); second chunk at K+16.
__device__ __forceinline__ v16bf load_frag(const bf16_t* p) {
    union { v16bf v; v8bf h[2]; } u;
    u.h[0] = *(const v8bf*)(p);
    u.h[1] = *(const v8bf*)(p + 16);
    return u.v;
}
__device__ __forceinline__ v16bf load_frag_lds(const char* p) {
    union { v16bf v; v8bf h[2]; } u;
    u.h[0] = *(const v8bf*)(p);
    u.h[1] = *(const v8bf*)(p + 32);   // +16 bf16 elements
    return u.v;
}

// ---------------- utility kernels ----------------
__global__ void zero_kernel(float* p, int n) {
    int i = blockIdx.x * 256 + threadIdx.x;
    if (i < n) p[i] = 0.0f;
}

// weights -> bf16, transposed to [ (p,s,o) ][ k ]
__global__ void wconv_kernel(const float* __restrict__ Wrel,
                             const float* __restrict__ Wroot,
                             bf16_t* __restrict__ wbf) {
    int idx = blockIdx.x * 256 + threadIdx.x;   // (((p*4+s)*256+o)*256+k)
    int k = idx & 255;
    int o = (idx >> 8) & 255;
    int s = (idx >> 16) & 3;
    int p = idx >> 18;
    float v;
    if (s < 3) v = Wrel[(((size_t)p * 3 + s) * 256 + k) * 256 + o];
    else       v = Wroot[((size_t)p * 256 + k) * 256 + o];
    wbf[idx] = (bf16_t)v;
}

// x = bf16(E[eids])
__global__ void gatherx_kernel(const float* __restrict__ E,
                               const int* __restrict__ eids,
                               bf16_t* __restrict__ xbf) {
    int idx = blockIdx.x * 256 + threadIdx.x;   // n*256 + k
    int n = idx >> 8, k = idx & 255;
    xbf[idx] = (bf16_t)E[(size_t)eids[n] * 256 + k];
}

// per-(dst,rel) in-degree counts
__global__ void count_kernel(const int* __restrict__ ei,
                             const int* __restrict__ rel,
                             float* __restrict__ cnt) {
    int e = blockIdx.x * 256 + threadIdx.x;
    int dst = ei[EEN + e];
    atomicAdd(&cnt[(size_t)dst * RNUM + rel[e]], 1.0f);
}

// ---------------- WMMA GEMM: xw[N][1024] = xbf[N][256] * Wt^T ----------------
// wave -> 16(M) x 64(cols) tile; 4 waves/block share one M tile.
// A tile (16x256 bf16, 8KB) staged into LDS once per block by the Tensor Data
// Mover (async DMA, TENSORcnt), with 16B/row padding for bank spread.
// B fragments are double-buffered in registers to pipeline global loads
// against the WMMAs. Slot 3 (cols 768..1023) accumulators start at bias[o].
__global__ __launch_bounds__(128) void gemm_kernel(const bf16_t* __restrict__ xbf,
                                                   const bf16_t* __restrict__ wt,   // [1024][256] this path
                                                   const float*  __restrict__ bias, // [256] this path
                                                   float* __restrict__ xw) {
    __shared__ __align__(16) char atile[16 * APITCH];

    const int lane = threadIdx.x & 31;
    const int wave = threadIdx.x >> 5;
    const int m0   = blockIdx.x * 16;
    const int cg   = blockIdx.y * 4 + wave;       // 0..15
    const int col0 = cg * 64;
    const int l15  = lane & 15;
    const int koff = (lane < 16) ? 0 : 8;
    const int slot = col0 >> 8;

#if __has_builtin(__builtin_amdgcn_tensor_load_to_lds)
    if (wave == 0) {
        const unsigned long long ga =
            (unsigned long long)(uintptr_t)(xbf + (size_t)m0 * 256);
        const unsigned int lb = (unsigned int)(uintptr_t)&atile[0];
        // D# group 0: count=1 | lds_addr | global_addr[56:0] | type=2
        v4u g0 = { 1u,
                   lb,
                   (unsigned int)(ga & 0xFFFFFFFFu),
                   (unsigned int)(((ga >> 32) & 0x1FFFFFFu) | 0x80000000u) };
        // D# group 1: data_size=2B, pad_enable, pad_interval=128dw, pad_amount=4dw,
        // tensor_dim0=256, tensor_dim1=60000, tile 256x16, dim0_stride=256
        v8i g1;
        g1[0] = (3 << 25) | (6 << 22) | (1 << 20) | (1 << 16);
        g1[1] = (int)(256u << 16);            // tensor_dim0 low16
        g1[2] = (int)((unsigned)NN << 16);    // tensor_dim0 hi16=0 | tensor_dim1 low16
        g1[3] = (int)(256u << 16);            // tensor_dim1 hi16=0 | tile_dim0=256
        g1[4] = 16;                           // tile_dim1=16 | tile_dim2=0
        g1[5] = 256;                          // tensor_dim0_stride low32
        g1[6] = 0;
        g1[7] = 0;
        v4i g2 = {0, 0, 0, 0};
        v4i g3 = {0, 0, 0, 0};
#if __has_include(<hip/amd_detail/amd_gfx1250_TDM.h>)
        v8i gz = {0, 0, 0, 0, 0, 0, 0, 0};
        __builtin_amdgcn_tensor_load_to_lds(g0, g1, g2, g3, gz, 0);
#else
        __builtin_amdgcn_tensor_load_to_lds(g0, g1, g2, g3, 0);
#endif
        __builtin_amdgcn_s_wait_tensorcnt(0);
    }
#else
    // cooperative fallback: copy A tile with the same padded pitch
    for (int c = threadIdx.x; c < 16 * 32; c += 128) {
        int row = c >> 5, chunk = c & 31;
        *(uint4*)&atile[row * APITCH + chunk * 16] =
            *(const uint4*)(xbf + (size_t)(m0 + row) * 256 + chunk * 8);
    }
#endif
    __syncthreads();

    v8f acc[4];
#pragma unroll
    for (int t = 0; t < 4; ++t) {
        float cinit = (slot == 3) ? bias[(col0 + t * 16 + l15) & 255] : 0.0f;
#pragma unroll
        for (int i = 0; i < 8; ++i) acc[t][i] = cinit;
    }

    const bf16_t* brow = wt + (size_t)(col0 + l15) * 256 + koff;

    v16bf bcur[4], bnxt[4];
#pragma unroll
    for (int t = 0; t < 4; ++t) bcur[t] = load_frag(brow + (size_t)t * 4096);

#pragma unroll
    for (int k0 = 0; k0 < 256; k0 += 32) {
        if (k0 + 32 < 256) {
#pragma unroll
            for (int t = 0; t < 4; ++t)
                bnxt[t] = load_frag(brow + (size_t)t * 4096 + k0 + 32);
        }
        v16bf a = load_frag_lds(&atile[l15 * APITCH + (k0 + koff) * 2]);
#pragma unroll
        for (int t = 0; t < 4; ++t)
            acc[t] = __builtin_amdgcn_wmma_f32_16x16x32_bf16(
                false, a, false, bcur[t], (short)0, acc[t], false, false);
#pragma unroll
        for (int t = 0; t < 4; ++t) bcur[t] = bnxt[t];
    }

    const int rbase = m0 + ((lane < 16) ? 0 : 8);
#pragma unroll
    for (int t = 0; t < 4; ++t) {
        const int c = col0 + t * 16 + l15;
#pragma unroll
        for (int v = 0; v < 8; ++v)
            xw[(size_t)(rbase + v) * CW + c] = acc[t][v];
    }
}

// ---------------- edge message scatter ----------------
// 64 threads per edge, float4 each: read xw[src][rel][j..], atomic-add into slot 3.
__global__ __launch_bounds__(256) void scatter_kernel(const int* __restrict__ ei,
                                                      const int* __restrict__ rel,
                                                      const float* __restrict__ cnt,
                                                      float* __restrict__ xw) {
    const int e = blockIdx.x * 4 + (threadIdx.x >> 6);
    const int j = (threadIdx.x & 63) * 4;
    const int src = ei[e];
    const int dst = ei[EEN + e];
    const int r   = rel[e];
    const float norm = 1.0f / fmaxf(cnt[(size_t)dst * RNUM + r], 1.0f);
    const float4 m = *(const float4*)(xw + ((size_t)src * SN + r) * OOUT + j);
    float* h = xw + ((size_t)dst * SN + 3) * OOUT + j;
    atomicAdd(h + 0, m.x * norm);
    atomicAdd(h + 1, m.y * norm);
    atomicAdd(h + 2, m.z * norm);
    atomicAdd(h + 3, m.w * norm);
}

// relu + pack into z[n][p][o], n < NREG
__global__ void reluz_kernel(const float* __restrict__ xw, float* __restrict__ z, int p) {
    int idx = blockIdx.x * 256 + threadIdx.x;   // n*256+o
    int n = idx >> 8, o = idx & 255;
    float v = xw[(size_t)n * CW + 3 * OOUT + o];
    z[((size_t)n * PNUM + p) * OOUT + o] = fmaxf(v, 0.0f);
}

// ---------------- semantic attention ----------------
// one block per (n,p) row: proj = tanh(z @ w1 + b1) @ w2, summed into psum[p]
__global__ __launch_bounds__(128) void attn_kernel(const float* __restrict__ z,
                                                   const float* __restrict__ w1,
                                                   const float* __restrict__ b1,
                                                   const float* __restrict__ w2,
                                                   float* __restrict__ psum) {
    __shared__ float zr[256];
    __shared__ float red[128];
    const int r = blockIdx.x;          // n*4+p
    const int tid = threadIdx.x;       // hidden unit h
    const float* zrow = z + (size_t)r * 256;
    zr[tid]       = zrow[tid];
    zr[tid + 128] = zrow[tid + 128];
    __syncthreads();
    float y = b1[tid];
#pragma unroll 8
    for (int d = 0; d < 256; ++d) y += zr[d] * w1[d * HH + tid];
    red[tid] = tanhf(y) * w2[tid];
    __syncthreads();
    for (int s = 64; s > 0; s >>= 1) {
        if (tid < s) red[tid] += red[tid + s];
        __syncthreads();
    }
    if (tid == 0) atomicAdd(&psum[r & 3], red[0]);
}

__global__ void beta_kernel(const float* __restrict__ psum, float* __restrict__ beta) {
    float m[4]; float mx = -1e30f;
    for (int p = 0; p < 4; ++p) { m[p] = psum[p] * (1.0f / (float)NREG); mx = fmaxf(mx, m[p]); }
    float s = 0.0f;
    for (int p = 0; p < 4; ++p) { m[p] = __expf(m[p] - mx); s += m[p]; }
    for (int p = 0; p < 4; ++p) beta[p] = m[p] / s;
}

__global__ void combine_kernel(const float* __restrict__ z,
                               const float* __restrict__ beta,
                               float* __restrict__ out) {
    int idx = blockIdx.x * 256 + threadIdx.x;   // n*256+o
    int n = idx >> 8, o = idx & 255;
    const float* zr = z + (size_t)n * (PNUM * OOUT) + o;
    out[idx] = beta[0] * zr[0] + beta[1] * zr[256] + beta[2] * zr[512] + beta[3] * zr[768];
}

// ---------------- host launcher ----------------
extern "C" void kernel_launch(void* const* d_in, const int* in_sizes, int n_in,
                              void* d_out, int out_size, void* d_ws, size_t ws_size,
                              hipStream_t stream) {
    // dict order: (edge_index_i, rel_i, eids_i) x4, E, ifdropout, W_rel, W_root,
    //             bias, att_w1, att_b1, att_w2
    const int* ei[PNUM]; const int* rel[PNUM]; const int* eid[PNUM];
    for (int i = 0; i < PNUM; ++i) {
        ei[i]  = (const int*)d_in[3 * i + 0];
        rel[i] = (const int*)d_in[3 * i + 1];
        eid[i] = (const int*)d_in[3 * i + 2];
    }
    const float* E      = (const float*)d_in[12];
    const float* W_rel  = (const float*)d_in[14];
    const float* W_root = (const float*)d_in[15];
    const float* bias   = (const float*)d_in[16];
    const float* aw1    = (const float*)d_in[17];
    const float* ab1    = (const float*)d_in[18];
    const float* aw2    = (const float*)d_in[19];
    float* out = (float*)d_out;

    // workspace carve-out (256B aligned regions)
    uintptr_t base = (uintptr_t)d_ws;
    auto carve = [&](size_t bytes) -> uintptr_t {
        uintptr_t r = base;
        base += (bytes + 255) & ~(size_t)255;
        return r;
    };
    bf16_t* wbf  = (bf16_t*)carve((size_t)PNUM * CW * DIN * sizeof(bf16_t)); // 2 MB
    bf16_t* xbf  = (bf16_t*)carve((size_t)NN * DIN * sizeof(bf16_t));        // 30.7 MB
    float*  xw   = (float*) carve((size_t)NN * CW * sizeof(float));          // 245.8 MB
    float*  cnt  = (float*) carve((size_t)NN * RNUM * sizeof(float));        // 0.7 MB
    float*  zbuf = (float*) carve((size_t)NREG * PNUM * OOUT * sizeof(float)); // 204.8 MB
    float*  psum = (float*) carve(64);
    float*  beta = (float*) carve(64);
    (void)ws_size; (void)in_sizes; (void)n_in; (void)out_size;

    // weights -> bf16 transposed (once per launch)
    wconv_kernel<<<(PNUM * CW * DIN) / 256, 256, 0, stream>>>(W_rel, W_root, wbf);
    zero_kernel<<<1, 256, 0, stream>>>(psum, 4);

    for (int p = 0; p < PNUM; ++p) {
        gatherx_kernel<<<(NN * DIN) / 256, 256, 0, stream>>>(E, eid[p], xbf);
        zero_kernel<<<(NN * RNUM + 255) / 256, 256, 0, stream>>>(cnt, NN * RNUM);
        count_kernel<<<EEN / 256, 256, 0, stream>>>(ei[p], rel[p], cnt);
        gemm_kernel<<<dim3(NN / 16, 4), 128, 0, stream>>>(
            xbf, wbf + (size_t)p * CW * DIN, bias + (size_t)p * OOUT, xw);
        scatter_kernel<<<EEN / 4, 256, 0, stream>>>(ei[p], rel[p], cnt, xw);
        reluz_kernel<<<(NREG * OOUT) / 256, 256, 0, stream>>>(xw, zbuf, p);
    }

    attn_kernel<<<NREG * PNUM, 128, 0, stream>>>(zbuf, aw1, ab1, aw2, psum);
    beta_kernel<<<1, 1, 0, stream>>>(psum, beta);
    combine_kernel<<<(NREG * OOUT) / 256, 256, 0, stream>>>(zbuf, beta, out);
}
